// QKNormMultiheadAttention_78529182040599
// MI455X (gfx1250) — matrix operand order repaired and checked by reading, not compiled, not run on hardware
//
#include <hip/hip_runtime.h>

// ---------------------------------------------------------------------------
// QK-Norm Multihead Attention for MI455X (gfx1250, wave32, WMMA bf16,
// async global->LDS staging for weight tiles)
// ---------------------------------------------------------------------------
#define D_MODEL  2048
#define N_HEADS  16
#define HEAD_DIM 128
#define SEQ      2048
#define BATCH    2
#define ROWS     (BATCH * SEQ)   // 4096

typedef __attribute__((ext_vector_type(16))) __bf16 v16bf;
typedef __attribute__((ext_vector_type(8)))  float  v8f;
typedef int v4i_gcc __attribute__((vector_size(16)));   // matches builtin proto

#define AS_GLOBAL __attribute__((address_space(1)))
#define AS_LDS    __attribute__((address_space(3)))

union ABReg { v16bf v; uint4 u[2]; };

#ifndef __has_builtin
#define __has_builtin(x) 0
#endif
#if __has_builtin(__builtin_amdgcn_global_load_async_to_lds_b128)
#define HAVE_ASYNC_LDS 1
#else
#define HAVE_ASYNC_LDS 0
#endif

__device__ __forceinline__ unsigned short f2bf(float f) {
  unsigned int u = __float_as_uint(f);
  u += 0x7FFFu + ((u >> 16) & 1u);           // round-to-nearest-even
  return (unsigned short)(u >> 16);
}

// --------------------------- fp32 -> bf16 convert ---------------------------
__global__ void cvt_bf16_kernel(const float* __restrict__ src,
                                unsigned short* __restrict__ dst, int n) {
  int i = blockIdx.x * blockDim.x + threadIdx.x;
  int stride = gridDim.x * blockDim.x;
  for (; i < n; i += stride) dst[i] = f2bf(src[i]);
}

// ------------------ cooperative 128x32 bf16 weight-tile stage ----------------
// thread tid (0..127) copies the 64B (32 bf16) of W row (col0 + tid) at k.
__device__ __forceinline__ void stage_b_tile(const unsigned short* __restrict__ wbase,
                                             unsigned short* lds_dst, int tid, int k) {
  const unsigned short* g = wbase + (size_t)tid * D_MODEL + k;
  unsigned short* l = lds_dst + tid * 32;
#if HAVE_ASYNC_LDS
  __builtin_amdgcn_global_load_async_to_lds_b128(
      (AS_GLOBAL v4i_gcc*)(g), (AS_LDS v4i_gcc*)(l), 0, 0);
  __builtin_amdgcn_global_load_async_to_lds_b128(
      (AS_GLOBAL v4i_gcc*)(g + 8), (AS_LDS v4i_gcc*)(l + 8), 0, 0);
  __builtin_amdgcn_global_load_async_to_lds_b128(
      (AS_GLOBAL v4i_gcc*)(g + 16), (AS_LDS v4i_gcc*)(l + 16), 0, 0);
  __builtin_amdgcn_global_load_async_to_lds_b128(
      (AS_GLOBAL v4i_gcc*)(g + 24), (AS_LDS v4i_gcc*)(l + 24), 0, 0);
#else
  const uint4* gv = reinterpret_cast<const uint4*>(g);
  uint4*       lv = reinterpret_cast<uint4*>(l);
  lv[0] = gv[0]; lv[1] = gv[1]; lv[2] = gv[2]; lv[3] = gv[3];
#endif
}

__device__ __forceinline__ void stage_wait() {
#if HAVE_ASYNC_LDS
# if __has_builtin(__builtin_amdgcn_s_wait_asynccnt)
  __builtin_amdgcn_s_wait_asynccnt(0);
# else
  asm volatile("s_wait_asynccnt 0x0" ::: "memory");
# endif
#endif
}

// ---------------- shared 64x128 GEMM core (B staged through LDS) -------------
// abase: 64 rows of row-major bf16 A (ld = D_MODEL)
// wbase: rows col0..col0+127 of row-major bf16 W (ld = D_MODEL)
// Bsh:   __shared__ 2 * 128*32 bf16 (double buffer)
__device__ __forceinline__ void gemm_tile_64x128(const unsigned short* __restrict__ abase,
                                                 const unsigned short* __restrict__ wbase,
                                                 unsigned short* Bsh, v8f acc[8]) {
  const int tid  = threadIdx.x;
  const int lane = tid & 31;
  const int wave = tid >> 5;
  const int n    = lane & 15;
  const int half = lane >> 4;

  const unsigned short* aptr = abase + (size_t)(wave * 16 + n) * D_MODEL + 8 * half;
  unsigned short* bufs[2] = { Bsh, Bsh + 128 * 32 };

  stage_b_tile(wbase, bufs[0], tid, 0);
  ABReg a_cur;
  a_cur.u[0] = *reinterpret_cast<const uint4*>(aptr);
  a_cur.u[1] = *reinterpret_cast<const uint4*>(aptr + 16);

  const int NI = D_MODEL / 32;
  for (int i = 0; i < NI; ++i) {
    unsigned short* cur = bufs[i & 1];
    unsigned short* nxt = bufs[(i + 1) & 1];

    stage_wait();          // staged data for `cur` complete (per-wave)
    __syncthreads();       // all waves staged & finished reading `nxt` last iter

    ABReg a_use = a_cur;
    if (i + 1 < NI) {
      stage_b_tile(wbase, nxt, tid, (i + 1) * 32);   // DMA overlaps compute below
      a_cur.u[0] = *reinterpret_cast<const uint4*>(aptr + (i + 1) * 32);
      a_cur.u[1] = *reinterpret_cast<const uint4*>(aptr + (i + 1) * 32 + 16);
    }

#pragma unroll
    for (int t = 0; t < 8; ++t) {
      ABReg b;
      const unsigned short* bp = cur + (t * 16 + n) * 32 + 8 * half;
      b.u[0] = *reinterpret_cast<const uint4*>(bp);
      b.u[1] = *reinterpret_cast<const uint4*>(bp + 16);
      acc[t] = __builtin_amdgcn_wmma_f32_16x16x32_bf16(
          false, a_use.v, false, b.v, (short)0, acc[t], false, false);
    }
  }
}

// --------------------- fused QKV projection + QK L2 norm --------------------
// grid: (ROWS/64, N_HEADS, 3)  block: 128 (4 waves)
// z=0 -> Qn (normalized * 1/sqrt(hd)) bf16 (B,H,L,hd)
// z=1 -> Kn (normalized)              bf16 (B,H,L,hd)
// z=2 -> Vt                           bf16 (B,H,hd,L)  (transposed)
__global__ __launch_bounds__(128)
void qkv_gemm_kernel(const unsigned short* __restrict__ xb,
                     const unsigned short* __restrict__ wq,
                     const unsigned short* __restrict__ wk,
                     const unsigned short* __restrict__ wv,
                     const float* __restrict__ bq,
                     const float* __restrict__ bk,
                     const float* __restrict__ bv,
                     unsigned short* __restrict__ qn,
                     unsigned short* __restrict__ kn,
                     unsigned short* __restrict__ vt) {
  __shared__ alignas(16) unsigned short Bsh[2 * 128 * 32];

  const int lane = threadIdx.x & 31;
  const int wave = threadIdx.x >> 5;
  const int n    = lane & 15;
  const int half = lane >> 4;
  const int h    = blockIdx.y;
  const int z    = blockIdx.z;

  const unsigned short* W    = (z == 0) ? wq : (z == 1) ? wk : wv;
  const float*          bias = (z == 0) ? bq : (z == 1) ? bk : bv;

  v8f acc[8];
#pragma unroll
  for (int t = 0; t < 8; ++t)
#pragma unroll
    for (int r = 0; r < 8; ++r) acc[t][r] = 0.0f;

  gemm_tile_64x128(xb + (size_t)blockIdx.x * 64 * D_MODEL,
                   W + (size_t)(h * HEAD_DIM) * D_MODEL, Bsh, acc);

  // bias (uniform per column)
#pragma unroll
  for (int t = 0; t < 8; ++t) {
    float bc = bias[h * HEAD_DIM + t * 16 + n];
#pragma unroll
    for (int r = 0; r < 8; ++r) acc[t][r] += bc;
  }

  const int b_idx   = (blockIdx.x * 64) >> 11;            // batch index
  const int seqbase = (blockIdx.x * 64 + wave * 16) & (SEQ - 1);

  if (z < 2) {
    // row-wise L2 norm over the full 128 head columns held by this wave
    float inv[8];
#pragma unroll
    for (int r = 0; r < 8; ++r) {
      float ss = 0.0f;
#pragma unroll
      for (int t = 0; t < 8; ++t) ss += acc[t][r] * acc[t][r];
      ss += __shfl_xor(ss, 1, 32);
      ss += __shfl_xor(ss, 2, 32);
      ss += __shfl_xor(ss, 4, 32);
      ss += __shfl_xor(ss, 8, 32);
      float s = 1.0f / (sqrtf(ss) + 1e-6f);
      if (z == 0) s *= 0.08838834764831845f;   // fold 1/sqrt(HEAD_DIM) into Q
      inv[r] = s;
    }
    unsigned short* dst = (z == 0) ? qn : kn;
    const size_t base = (size_t)(b_idx * N_HEADS + h) * SEQ;
#pragma unroll
    for (int r = 0; r < 8; ++r) {
      int seq = seqbase + r + 8 * half;
      unsigned short* p = dst + (base + seq) * HEAD_DIM + n;
#pragma unroll
      for (int t = 0; t < 8; ++t) p[t * 16] = f2bf(acc[t][r] * inv[r]);
    }
  } else {
    // V transposed: vt[(bh*128 + d)*SEQ + seq]; per lane d fixed, seq contiguous
    const size_t base = (size_t)(b_idx * N_HEADS + h) * HEAD_DIM;
#pragma unroll
    for (int t = 0; t < 8; ++t) {
      int d = t * 16 + n;
      unsigned short* p = vt + (base + d) * SEQ + seqbase + 8 * half;
#pragma unroll
      for (int r = 0; r < 8; ++r) p[r] = f2bf(acc[t][r]);
    }
  }
}

// ----------------------------- flash attention ------------------------------
// grid: (SEQ/64, N_HEADS, BATCH)  block: 128 (4 waves), wave = 16 query rows
__global__ __launch_bounds__(128)
void attention_kernel(const unsigned short* __restrict__ qn,
                      const unsigned short* __restrict__ kn,
                      const unsigned short* __restrict__ vt,
                      unsigned short* __restrict__ attn) {
  __shared__ alignas(16) unsigned short pLds[4][16 * 32];  // per-wave private

  const int lane = threadIdx.x & 31;
  const int wave = threadIdx.x >> 5;
  const int n    = lane & 15;
  const int half = lane >> 4;
  const int b    = blockIdx.z;
  const int h    = blockIdx.y;
  const int bh   = b * N_HEADS + h;
  const int qbase = blockIdx.x * 64 + wave * 16;

  const unsigned short* Q = qn + (size_t)bh * SEQ * HEAD_DIM;
  const unsigned short* K = kn + (size_t)bh * SEQ * HEAD_DIM;
  const unsigned short* V = vt + (size_t)bh * HEAD_DIM * SEQ;

  // Q as A operand, 4 chunks of K=32 covering head_dim=128 (register-resident)
  ABReg qa[4];
  {
    const unsigned short* qp = Q + (size_t)(qbase + n) * HEAD_DIM + 8 * half;
#pragma unroll
    for (int c = 0; c < 4; ++c) {
      qa[c].u[0] = *reinterpret_cast<const uint4*>(qp + c * 32);
      qa[c].u[1] = *reinterpret_cast<const uint4*>(qp + c * 32 + 16);
    }
  }

  v8f oacc[8];
#pragma unroll
  for (int t = 0; t < 8; ++t)
#pragma unroll
    for (int r = 0; r < 8; ++r) oacc[t][r] = 0.0f;

  float mrow[8], lrow[8];
#pragma unroll
  for (int r = 0; r < 8; ++r) { mrow[r] = -1e30f; lrow[r] = 0.0f; }

  unsigned short* myP = pLds[wave];

  for (int kt = 0; kt < SEQ; kt += 32) {
    // ---- batch-load all K B-operands (one clause, staggered waits) ----
    const unsigned short* kp0 = K + (size_t)(kt + n) * HEAD_DIM + 8 * half;
    const unsigned short* kp1 = kp0 + 16 * HEAD_DIM;
    ABReg kb[8];
#pragma unroll
    for (int c = 0; c < 4; ++c) {
      kb[2 * c].u[0]     = *reinterpret_cast<const uint4*>(kp0 + c * 32);
      kb[2 * c].u[1]     = *reinterpret_cast<const uint4*>(kp0 + c * 32 + 16);
      kb[2 * c + 1].u[0] = *reinterpret_cast<const uint4*>(kp1 + c * 32);
      kb[2 * c + 1].u[1] = *reinterpret_cast<const uint4*>(kp1 + c * 32 + 16);
    }

    // ---- S = Q * K^T (two 16x16 tiles: keys kt..+15, kt+16..+31) ----
    v8f s0, s1;
#pragma unroll
    for (int r = 0; r < 8; ++r) { s0[r] = 0.0f; s1[r] = 0.0f; }
#pragma unroll
    for (int c = 0; c < 4; ++c) {
      s0 = __builtin_amdgcn_wmma_f32_16x16x32_bf16(false, qa[c].v, false, kb[2 * c].v,
                                                   (short)0, s0, false, false);
      s1 = __builtin_amdgcn_wmma_f32_16x16x32_bf16(false, qa[c].v, false, kb[2 * c + 1].v,
                                                   (short)0, s1, false, false);
    }

    // ---- issue V B-operand loads early: VMEM overlaps softmax VALU work ----
    ABReg vb[8];
    const unsigned short* vbase = V + kt + 8 * half;
#pragma unroll
    for (int t = 0; t < 8; ++t) {
      const unsigned short* vp = vbase + (size_t)(t * 16 + n) * SEQ;
      vb[t].u[0] = *reinterpret_cast<const uint4*>(vp);
      vb[t].u[1] = *reinterpret_cast<const uint4*>(vp + 16);
    }

    // ---- online softmax (row stats replicated across the 16 N-lanes) ----
    float alpha[8];
#pragma unroll
    for (int r = 0; r < 8; ++r) {
      float mx = fmaxf(s0[r], s1[r]);
      mx = fmaxf(mx, __shfl_xor(mx, 1, 32));
      mx = fmaxf(mx, __shfl_xor(mx, 2, 32));
      mx = fmaxf(mx, __shfl_xor(mx, 4, 32));
      mx = fmaxf(mx, __shfl_xor(mx, 8, 32));
      float mnew = fmaxf(mrow[r], mx);
      alpha[r] = __expf(mrow[r] - mnew);
      float p0 = __expf(s0[r] - mnew);
      float p1 = __expf(s1[r] - mnew);
      float ls = p0 + p1;
      ls += __shfl_xor(ls, 1, 32);
      ls += __shfl_xor(ls, 2, 32);
      ls += __shfl_xor(ls, 4, 32);
      ls += __shfl_xor(ls, 8, 32);
      lrow[r] = lrow[r] * alpha[r] + ls;
      mrow[r] = mnew;
      // spill P (D layout: row r+8*half, col = tile*16 + n) to LDS 16x32
      int rowoff = (r + 8 * half) * 32;
      myP[rowoff + n]      = f2bf(p0);
      myP[rowoff + 16 + n] = f2bf(p1);
    }

    // rescale running O accumulators
#pragma unroll
    for (int t = 0; t < 8; ++t)
#pragma unroll
      for (int r = 0; r < 8; ++r) oacc[t][r] *= alpha[r];

    // same-wave cross-lane LDS RAW: drain DS pipe (wave-synchronous, no barrier)
    asm volatile("s_wait_dscnt 0x0" ::: "memory");

    // re-read P in A-operand layout (16 rows x K=32)
    ABReg pa;
    {
      const unsigned short* pp = myP + n * 32 + 8 * half;
      pa.u[0] = *reinterpret_cast<const uint4*>(pp);
      pa.u[1] = *reinterpret_cast<const uint4*>(pp + 16);
    }

    // ---- O += P * V ----
#pragma unroll
    for (int t = 0; t < 8; ++t) {
      oacc[t] = __builtin_amdgcn_wmma_f32_16x16x32_bf16(
          false, pa.v, false, vb[t].v, (short)0, oacc[t], false, false);
    }
  }

  // epilogue: O /= l, store bf16 into (B*L, D) row-major head slice
#pragma unroll
  for (int r = 0; r < 8; ++r) {
    float invl = 1.0f / lrow[r];
    int qrow = qbase + r + 8 * half;
    unsigned short* p = attn + ((size_t)(b * SEQ + qrow)) * D_MODEL + h * HEAD_DIM + n;
#pragma unroll
    for (int t = 0; t < 8; ++t) p[t * 16] = f2bf(oacc[t][r] * invl);
  }
}

// ----------------------------- output projection ----------------------------
// grid: (ROWS/64, D_MODEL/128)  block: 128
__global__ __launch_bounds__(128)
void out_proj_kernel(const unsigned short* __restrict__ ab,
                     const unsigned short* __restrict__ wo,
                     const float* __restrict__ bo,
                     float* __restrict__ out) {
  __shared__ alignas(16) unsigned short Bsh[2 * 128 * 32];

  const int lane = threadIdx.x & 31;
  const int wave = threadIdx.x >> 5;
  const int n    = lane & 15;
  const int half = lane >> 4;
  const int col0 = blockIdx.y * 128;
  const int rowbase = blockIdx.x * 64 + wave * 16;

  v8f acc[8];
#pragma unroll
  for (int t = 0; t < 8; ++t)
#pragma unroll
    for (int r = 0; r < 8; ++r) acc[t][r] = 0.0f;

  gemm_tile_64x128(ab + (size_t)blockIdx.x * 64 * D_MODEL,
                   wo + (size_t)col0 * D_MODEL, Bsh, acc);

#pragma unroll
  for (int t = 0; t < 8; ++t) {
    float bc = bo[col0 + t * 16 + n];
#pragma unroll
    for (int r = 0; r < 8; ++r) {
      int row = rowbase + r + 8 * half;
      out[(size_t)row * D_MODEL + col0 + t * 16 + n] = acc[t][r] + bc;
    }
  }
}

// ------------------------------- launcher -----------------------------------
extern "C" void kernel_launch(void* const* d_in, const int* in_sizes, int n_in,
                              void* d_out, int out_size, void* d_ws, size_t ws_size,
                              hipStream_t stream) {
  const float* x  = (const float*)d_in[0];
  const float* Wq = (const float*)d_in[1];
  const float* bq = (const float*)d_in[2];
  const float* Wk = (const float*)d_in[3];
  const float* bk = (const float*)d_in[4];
  const float* Wv = (const float*)d_in[5];
  const float* bv = (const float*)d_in[6];
  const float* Wo = (const float*)d_in[7];
  const float* bo = (const float*)d_in[8];
  float* out = (float*)d_out;

  char* ws = (char*)d_ws;
  size_t off = 0;
  const size_t XN = (size_t)ROWS * D_MODEL;          // 8.4M elements
  const size_t WN = (size_t)D_MODEL * D_MODEL;       // 4.2M elements

  unsigned short* xb  = (unsigned short*)(ws + off); off += XN * 2;
  unsigned short* wqb = (unsigned short*)(ws + off); off += WN * 2;
  unsigned short* wkb = (unsigned short*)(ws + off); off += WN * 2;
  unsigned short* wvb = (unsigned short*)(ws + off); off += WN * 2;
  unsigned short* wob = (unsigned short*)(ws + off); off += WN * 2;
  unsigned short* qn  = (unsigned short*)(ws + off); off += XN * 2;
  unsigned short* kn  = (unsigned short*)(ws + off); off += XN * 2;
  unsigned short* vt  = (unsigned short*)(ws + off); off += XN * 2;
  unsigned short* ao  = (unsigned short*)(ws + off); off += XN * 2;

  // fp32 -> bf16 conversions
  cvt_bf16_kernel<<<2048, 256, 0, stream>>>(x,  xb,  (int)XN);
  cvt_bf16_kernel<<<2048, 256, 0, stream>>>(Wq, wqb, (int)WN);
  cvt_bf16_kernel<<<2048, 256, 0, stream>>>(Wk, wkb, (int)WN);
  cvt_bf16_kernel<<<2048, 256, 0, stream>>>(Wv, wvb, (int)WN);
  cvt_bf16_kernel<<<2048, 256, 0, stream>>>(Wo, wob, (int)WN);

  // fused QKV projection + QK normalization (+ V transpose)
  qkv_gemm_kernel<<<dim3(ROWS / 64, N_HEADS, 3), 128, 0, stream>>>(
      xb, wqb, wkb, wvb, bq, bk, bv, qn, kn, vt);

  // flash attention
  attention_kernel<<<dim3(SEQ / 64, N_HEADS, BATCH), 128, 0, stream>>>(
      qn, kn, vt, ao);

  // output projection
  out_proj_kernel<<<dim3(ROWS / 64, D_MODEL / 128), 128, 0, stream>>>(
      ao, wob, bo, out);

  (void)in_sizes; (void)n_in; (void)out_size; (void)ws_size;
}
